// BiLstmCrf_4148938408766
// MI455X (gfx1250) — matrix operand order repaired and checked by reading, not compiled, not run on hardware
//
#include <hip/hip_runtime.h>
#include <math.h>

// ---------------------------------------------------------------------------
// BiLSTM-CRF forward for MI455X (gfx1250, wave32, WMMA + async-to-LDS DMA).
// bf16 WMMA (f32 accum) for all GEMMs; f32 elementwise/CRF.
// ---------------------------------------------------------------------------

typedef __attribute__((ext_vector_type(16))) __bf16 v16bf;
typedef __attribute__((ext_vector_type(8)))  float  v8f;

union V16 { v16bf v; uint4 q[2]; unsigned short u[16]; };

constexpr int Bn = 256;   // batch
constexpr int Tn = 512;   // time
constexpr int En = 256;   // embed dim
constexpr int Hd = 128;   // per-direction hidden
constexpr int G4 = 512;   // 4*Hd gates
constexpr int Cn = 30;    // tags
constexpr int BT = Bn * Tn;

static __device__ __forceinline__ unsigned short f2bf(float f) {
  unsigned u = __float_as_uint(f);
  unsigned r = ((u >> 16) & 1u) + 0x7FFFu;          // round-to-nearest-even
  return (unsigned short)((u + r) >> 16);
}
static __device__ __forceinline__ float sigf(float x) { return 1.0f / (1.0f + expf(-x)); }

// LDS byte offset of a generic shared pointer (low 32 bits of the LDS aperture
// address are the workgroup-relative byte offset on gfx1250).
static __device__ __forceinline__ unsigned ldsOff(const void* p) {
  return (unsigned)(unsigned long long)(uintptr_t)p;
}
// CDNA5 async DMA: 16B global -> LDS, tracked by ASYNCcnt.
static __device__ __forceinline__ void asyncLdsB128(unsigned ldsoff, const void* gptr) {
  asm volatile("global_load_async_to_lds_b128 %0, %1, off"
               :: "v"(ldsoff), "v"((unsigned long long)(uintptr_t)gptr)
               : "memory");
}
static __device__ __forceinline__ void waitAsync0() {
  asm volatile("s_wait_asynccnt 0x0" ::: "memory");
}
static __device__ __forceinline__ void waitAsync8() {
  asm volatile("s_wait_asynccnt 0x8" ::: "memory");
}

// --------------------------- zero scalar output ----------------------------
__global__ void k_zero(float* o) { if (threadIdx.x == 0) o[0] = 0.0f; }

// --------------------------- f32 -> bf16 convert ---------------------------
__global__ void k_f2bf(const float* __restrict__ in, unsigned short* __restrict__ out, int n) {
  int i = blockIdx.x * 256 + threadIdx.x;
  if (i < n) out[i] = f2bf(in[i]);
}

// ------------------- embedding gather + bf16 conversion --------------------
__global__ __launch_bounds__(256) void k_embed(const int* __restrict__ chars,
                                               const float* __restrict__ embed,
                                               unsigned short* __restrict__ xbf) {
  size_t idx = (size_t)blockIdx.x * 256 + threadIdx.x;   // 0 .. BT*E/8-1
  size_t e8  = idx * 8;
  int row = (int)(e8 >> 8);       // /E
  int e0  = (int)(e8 & (En - 1));
  int ch  = chars[row];
  const float* src = embed + (size_t)ch * En + e0;
  float4 f0 = *(const float4*)(src);
  float4 f1 = *(const float4*)(src + 4);
  union { unsigned short u[8]; uint4 q; } o;
  o.u[0] = f2bf(f0.x); o.u[1] = f2bf(f0.y); o.u[2] = f2bf(f0.z); o.u[3] = f2bf(f0.w);
  o.u[4] = f2bf(f1.x); o.u[5] = f2bf(f1.y); o.u[6] = f2bf(f1.z); o.u[7] = f2bf(f1.w);
  *(uint4*)(xbf + e8) = o.q;
}

// -------------------- input projection GEMM (WMMA bf16) --------------------
// gx[row][g] = sum_e x_bf[row][e] * W[g][e] + bias[g]
// Block: 8 waves x (16M x 64N). All waves share one 64x256 bf16 W block,
// staged into LDS once via async DMA. grid=(1024, 8).
__global__ __launch_bounds__(256) void k_gemm_proj(const unsigned short* __restrict__ xbf,
                                                   const unsigned short* __restrict__ wbf,
                                                   const float* __restrict__ bias,
                                                   float* __restrict__ gx) {
  __shared__ unsigned short sB[64 * En];   // 32 KB weight block [col][k]
  const int tid = threadIdx.x;
  const int wave = tid >> 5, lane = tid & 31;
  const int mTile = blockIdx.x * 8 + wave;       // 0..8191
  const int nBase = blockIdx.y * 64;
  const int row0  = mTile * 16;
  const int m     = lane & 15;
  const int kSel  = lane >> 4;

  // Async-stage W block: 2048 x 16B chunks, 8 per thread.
  {
    const unsigned sBoff = ldsOff(sB);
#pragma unroll
    for (int j = 0; j < 8; ++j) {
      const int i = tid + j * 256;           // chunk id
      const int col = i >> 5;                // 32 chunks per 512B row
      const int o = i & 31;
      asyncLdsB128(sBoff + i * 16, wbf + (size_t)(nBase + col) * En + o * 8);
    }
  }
  waitAsync0();
  __syncthreads();

  const unsigned short* Arow = xbf + (size_t)(row0 + m) * En;
  v8f acc[4] = {};

  for (int k = 0; k < En; k += 32) {
    V16 a;  // A 16x32 bf16: per lane, halfs [k+kSel*8 .. +7] and [k+16+kSel*8 .. +7]
    a.q[0] = *(const uint4*)(Arow + k + kSel * 8);
    a.q[1] = *(const uint4*)(Arow + k + 16 + kSel * 8);
#pragma unroll
    for (int tN = 0; tN < 4; ++tN) {
      const unsigned short* bp = sB + (size_t)(tN * 16 + m) * En + k + kSel * 8;
      V16 bm;  // B 32x16 bf16 from LDS: K [k+kSel*8 .. +7] and [k+16+kSel*8 .. +7] of column
      bm.q[0] = *(const uint4*)(bp);
      bm.q[1] = *(const uint4*)(bp + 16);
      acc[tN] = __builtin_amdgcn_wmma_f32_16x16x32_bf16(
          false, a.v, false, bm.v, (short)0, acc[tN], false, false);
    }
  }
#pragma unroll
  for (int tN = 0; tN < 4; ++tN) {
    const int col = nBase + tN * 16 + m;
    const float bv = bias[col];
#pragma unroll
    for (int r = 0; r < 8; ++r) {   // C layout: VGPR r -> row r (+8 for upper lanes)
      const int rowm = row0 + r + kSel * 8;
      gx[(size_t)rowm * G4 + col] = acc[tN][r] + bv;
    }
  }
}

// --------------------- persistent bidirectional LSTM -----------------------
// grid=(B/16, 2): x=batch tile of 16 rows, y=direction. 8 waves/WG.
// LDS: W_hh 128K | hA 4K | gates 32K | gx double-buffer 64K | slen = ~228 KB.
// gx for step t+1 is DMA'd (async-to-LDS) while step t's WMMA runs.
constexpr unsigned WHH_OFF  = 0;
constexpr unsigned HA_OFF   = 131072;
constexpr unsigned GATE_OFF = 135168;
constexpr unsigned GX_OFF   = 167936;
constexpr unsigned GX_B     = 32768;
constexpr unsigned SLEN_OFF = 233472;
constexpr unsigned LSTM_LDS = 233536;

__global__ __launch_bounds__(256) void k_lstm(const float* __restrict__ gxF,
                                              const float* __restrict__ gxB,
                                              const unsigned short* __restrict__ whhF,
                                              const unsigned short* __restrict__ whhB,
                                              const int* __restrict__ seqlen,
                                              float* __restrict__ hF,
                                              float* __restrict__ hBr) {
  extern __shared__ char smem[];
  unsigned short* sWhh  = (unsigned short*)(smem + WHH_OFF);   // [G4][Hd] bf16
  unsigned short* sHa   = (unsigned short*)(smem + HA_OFF);    // [16][Hd] bf16
  float*          sGate = (float*)(smem + GATE_OFF);           // [16][G4] f32
  int*            slenS = (int*)(smem + SLEN_OFF);             // [16]
  const unsigned  smemBase = ldsOff(smem);

  const int dir = blockIdx.y;
  const float* gx = dir ? gxB : gxF;
  const unsigned short* whh = dir ? whhB : whhF;
  float* hOut = dir ? hBr : hF;
  const int bBase = blockIdx.x * 16;
  const int tid = threadIdx.x, lane = tid & 31, wave = tid >> 5;

  // Stage W_hh into LDS via async DMA (8192 x 16B) and zero the h tile.
#pragma unroll
  for (int j = 0; j < 32; ++j) {
    const int i = tid + j * 256;
    asyncLdsB128(smemBase + WHH_OFF + i * 16, (const char*)whh + i * 16);
  }
  for (int i = tid; i < (16 * Hd * 2) / 16; i += 256)
    ((uint4*)sHa)[i] = uint4{0u, 0u, 0u, 0u};
  if (tid < 16) slenS[tid] = seqlen[bBase + tid];

  // Per-thread cell state: pair p -> (m = 2p + tid/128, hd = tid%128)
  float c[8];
  int   len_[8];
#pragma unroll
  for (int p = 0; p < 8; ++p) {
    c[p] = 0.0f;
    len_[p] = seqlen[bBase + 2 * p + (tid >> 7)];
  }

  // Prefetch gx tile for t=0 into buffer 0 (2048 x 16B, 8 per thread).
  {
#pragma unroll
    for (int j = 0; j < 8; ++j) {
      const int i = tid + j * 256;          // 0..2047
      const int m = i >> 7, o16 = i & 127;  // row m, uint4 index within 512 floats
      int grow;
      if (dir == 0) {
        grow = (bBase + m) * Tn + 0;
      } else {
        int rt = slenS[m] - 1;
        rt = rt < 0 ? 0 : rt;
        grow = (bBase + m) * Tn + rt;
      }
      asyncLdsB128(smemBase + GX_OFF + (m << 11) + (o16 << 4),
                   gx + (size_t)grow * G4 + o16 * 4);
    }
  }
  waitAsync0();
  __syncthreads();

  const int nB = wave * 64;
  const int m16 = lane & 15, kSel = lane >> 4;
  const int hd = tid & 127;

  for (int t = 0; t < Tn; ++t) {
    const int nxt = t + 1;
    // ---- issue async DMA for next step's gx tile (overlaps the WMMAs) ----
    if (nxt < Tn) {
      const unsigned bufOff = GX_OFF + (unsigned)(nxt & 1) * GX_B;
#pragma unroll
      for (int j = 0; j < 8; ++j) {
        const int i = tid + j * 256;
        const int m = i >> 7, o16 = i & 127;
        int grow;
        if (dir == 0) {
          grow = (bBase + m) * Tn + nxt;
        } else {
          int rt = slenS[m] - 1 - nxt;
          rt = rt < 0 ? 0 : rt;
          grow = (bBase + m) * Tn + rt;
        }
        asyncLdsB128(smemBase + bufOff + (m << 11) + (o16 << 4),
                     gx + (size_t)grow * G4 + o16 * 4);
      }
    }

    // ---- phase 1: gates_pre = h @ W_hh^T via WMMA (16x64 tile per wave) ----
    v8f acc[4] = {};
#pragma unroll
    for (int k = 0; k < Hd; k += 32) {
      V16 a;
      const unsigned short* ap = sHa + m16 * Hd + k + kSel * 8;
      a.q[0] = *(const uint4*)(ap);
      a.q[1] = *(const uint4*)(ap + 16);
#pragma unroll
      for (int tN = 0; tN < 4; ++tN) {
        const unsigned short* bp = sWhh + (nB + tN * 16 + m16) * Hd + k + kSel * 8;
        V16 bm;
        bm.q[0] = *(const uint4*)(bp);
        bm.q[1] = *(const uint4*)(bp + 16);
        acc[tN] = __builtin_amdgcn_wmma_f32_16x16x32_bf16(
            false, a.v, false, bm.v, (short)0, acc[tN], false, false);
      }
    }
#pragma unroll
    for (int tN = 0; tN < 4; ++tN)
#pragma unroll
      for (int r = 0; r < 8; ++r)
        sGate[(r + kSel * 8) * G4 + nB + tN * 16 + m16] = acc[tN][r];

    // Oldest 8 async loads (this step's gx) must have landed; the 8 just
    // issued for t+1 may still be in flight (async completes in order).
    if (nxt < Tn) waitAsync8(); else waitAsync0();
    __syncthreads();

    // ---- phase 2: gates + gx(LDS) -> c,h update; write bf16 h to A-tile ---
    const float* gxl = (const float*)(smem + GX_OFF + (size_t)(t & 1) * GX_B);
#pragma unroll
    for (int p = 0; p < 8; ++p) {
      const int m = 2 * p + (tid >> 7);
      const int b = bBase + m;
      const float gi = sGate[m * G4 + hd]          + gxl[m * G4 + hd];
      const float gf = sGate[m * G4 + Hd + hd]     + gxl[m * G4 + Hd + hd];
      const float gg = sGate[m * G4 + 2 * Hd + hd] + gxl[m * G4 + 2 * Hd + hd];
      const float go = sGate[m * G4 + 3 * Hd + hd] + gxl[m * G4 + 3 * Hd + hd];
      const float cn = sigf(gf) * c[p] + sigf(gi) * tanhf(gg);
      const float hv = sigf(go) * tanhf(cn);
      c[p] = cn;
      hOut[(size_t)(b * Tn + t) * Hd + hd] = hv;
      sHa[m * Hd + hd] = f2bf(hv);
    }
    (void)len_;
    __syncthreads();
  }
}

// ------------------ fc + log_softmax + gold-emission gather ----------------
__global__ __launch_bounds__(256) void k_fc(const float* __restrict__ hF,
                                            const float* __restrict__ hBr,
                                            const int* __restrict__ seqlen,
                                            const int* __restrict__ target,
                                            const float* __restrict__ fcW,
                                            const float* __restrict__ fcb,
                                            float* __restrict__ logits,
                                            float* __restrict__ emitg) {
  __shared__ float sW[Cn * 258];   // padded stride 258 -> conflict-free rows
  __shared__ float sF[8][256];
  const int tid = threadIdx.x, lane = tid & 31, wave = tid >> 5;
  for (int i = tid; i < Cn * 256; i += 256)
    sW[(i >> 8) * 258 + (i & 255)] = fcW[i];

  const int pos = blockIdx.x * 8 + wave;        // b*T + t
  const int b = pos >> 9, t = pos & (Tn - 1);
  const int len = seqlen[b];
  int rt = len - 1 - t;
  rt = rt < 0 ? 0 : (rt > Tn - 1 ? Tn - 1 : rt);
  const float* hf = hF + (size_t)pos * Hd;
  const float* hb = hBr + (size_t)(b * Tn + rt) * Hd;   // un-reverse
  for (int i = lane; i < Hd; i += 32) {
    sF[wave][i] = hf[i];
    sF[wave][Hd + i] = hb[i];
  }
  __syncthreads();

  const int cIdx = lane;
  float v = -INFINITY;
  if (cIdx < Cn) {
    float s = 0.0f;
    const float* wr = sW + cIdx * 258;
#pragma unroll 8
    for (int k = 0; k < 256; ++k) s += sF[wave][k] * wr[k];
    v = s + fcb[cIdx];
  }
  float mx = v;
#pragma unroll
  for (int o = 16; o; o >>= 1) mx = fmaxf(mx, __shfl_xor(mx, o, 32));
  float ex = (cIdx < Cn) ? expf(v - mx) : 0.0f;
#pragma unroll
  for (int o = 16; o; o >>= 1) ex += __shfl_xor(ex, o, 32);
  const float lg = v - (mx + logf(ex));
  if (cIdx < Cn) logits[(size_t)pos * Cn + cIdx] = lg;
  const int tgt = target[pos];
  if (cIdx == tgt) emitg[pos] = lg;
}

// ----------------- CRF forward scan + gold score + mean --------------------
__global__ __launch_bounds__(256) void k_crf(const float* __restrict__ logits,
                                             const float* __restrict__ emitg,
                                             const int* __restrict__ target,
                                             const int* __restrict__ seqlen,
                                             const float* __restrict__ trans,
                                             const float* __restrict__ startv,
                                             const float* __restrict__ endv,
                                             float* __restrict__ out) {
  __shared__ float sT[Cn * Cn];
  const int tid = threadIdx.x, lane = tid & 31, wave = tid >> 5;
  for (int i = tid; i < Cn * Cn; i += 256) sT[i] = trans[i];
  __syncthreads();

  const int b = blockIdx.x * 8 + wave;
  const float* lg = logits + (size_t)b * Tn * Cn;
  const int len = seqlen[b];
  const int cIdx = lane;
  const int cc = cIdx < Cn ? cIdx : 0;

  float alpha = (cIdx < Cn) ? lg[cIdx] + startv[cIdx] : -INFINITY;
  for (int t = 1; t < len; ++t) {
    const float* em = lg + t * Cn;
    float mx = -INFINITY;
    for (int j = 0; j < Cn; ++j) {
      float v = __shfl(alpha, j, 32) + sT[j * Cn + cc];
      mx = fmaxf(mx, v);
    }
    float s = 0.0f;
    for (int j = 0; j < Cn; ++j) {
      float v = __shfl(alpha, j, 32) + sT[j * Cn + cc];
      s += expf(v - mx);
    }
    alpha = mx + logf(s) + em[cc];
  }
  float av = (cIdx < Cn) ? alpha + endv[cIdx] : -INFINITY;
  float mx = av;
#pragma unroll
  for (int o = 16; o; o >>= 1) mx = fmaxf(mx, __shfl_xor(mx, o, 32));
  float s = (cIdx < Cn) ? expf(av - mx) : 0.0f;
#pragma unroll
  for (int o = 16; o; o >>= 1) s += __shfl_xor(s, o, 32);
  const float norm = mx + logf(s);

  const int* tg = target + b * Tn;
  float g = 0.0f;
  for (int t = lane; t < len; t += 32) g += emitg[b * Tn + t];
  for (int t = 1 + lane; t < len; t += 32) g += sT[tg[t - 1] * Cn + tg[t]];
#pragma unroll
  for (int o = 16; o; o >>= 1) g += __shfl_xor(g, o, 32);
  if (lane == 0) {
    g += startv[tg[0]] + endv[tg[len - 1]];
    atomicAdd(out, (norm - g) * (1.0f / (float)Bn));
  }
}

// ---------------------------------------------------------------------------
extern "C" void kernel_launch(void* const* d_in, const int* in_sizes, int n_in,
                              void* d_out, int out_size, void* d_ws, size_t ws_size,
                              hipStream_t stream) {
  const int*   chars  = (const int*)d_in[0];
  const int*   target = (const int*)d_in[1];
  const int*   seqlen = (const int*)d_in[2];
  const float* embed  = (const float*)d_in[3];
  const float* Wihf   = (const float*)d_in[4];
  const float* Whhf   = (const float*)d_in[5];
  const float* bfv    = (const float*)d_in[6];
  const float* Wihb   = (const float*)d_in[7];
  const float* Whhb   = (const float*)d_in[8];
  const float* bbv    = (const float*)d_in[9];
  const float* fcW    = (const float*)d_in[10];
  const float* fcb    = (const float*)d_in[11];
  const float* trans  = (const float*)d_in[12];
  const float* startv = (const float*)d_in[13];
  const float* endv   = (const float*)d_in[14];
  float* out = (float*)d_out;
  (void)in_sizes; (void)n_in; (void)out_size; (void)ws_size;

  char* ws = (char*)d_ws;
  size_t off = 0;
  auto wsalloc = [&](size_t bytes) -> char* {
    char* p = ws + off;
    off += (bytes + 255) & ~(size_t)255;
    return p;
  };
  unsigned short* xbf  = (unsigned short*)wsalloc((size_t)BT * En * 2);   //  64 MB
  float* gxf           = (float*)wsalloc((size_t)BT * G4 * 4);            // 256 MB
  float* gxb           = (float*)wsalloc((size_t)BT * G4 * 4);            // 256 MB
  float* hF            = (float*)wsalloc((size_t)BT * Hd * 4);            //  64 MB
  float* hBr           = (float*)wsalloc((size_t)BT * Hd * 4);            //  64 MB
  unsigned short* wihf = (unsigned short*)wsalloc((size_t)G4 * En * 2);
  unsigned short* wihb = (unsigned short*)wsalloc((size_t)G4 * En * 2);
  unsigned short* whhf = (unsigned short*)wsalloc((size_t)G4 * Hd * 2);
  unsigned short* whhb = (unsigned short*)wsalloc((size_t)G4 * Hd * 2);
  float* logits        = (float*)wsalloc((size_t)BT * Cn * 4);            //  15 MB
  float* emitg         = (float*)wsalloc((size_t)BT * 4);

  k_zero<<<1, 32, 0, stream>>>(out);

  k_f2bf<<<(G4 * En + 255) / 256, 256, 0, stream>>>(Wihf, wihf, G4 * En);
  k_f2bf<<<(G4 * En + 255) / 256, 256, 0, stream>>>(Wihb, wihb, G4 * En);
  k_f2bf<<<(G4 * Hd + 255) / 256, 256, 0, stream>>>(Whhf, whhf, G4 * Hd);
  k_f2bf<<<(G4 * Hd + 255) / 256, 256, 0, stream>>>(Whhb, whhb, G4 * Hd);

  k_embed<<<(BT * En / 8) / 256, 256, 0, stream>>>(chars, embed, xbf);

  dim3 gg(1024, 8);  // 8192 M-tiles x 8 N-groups(64)
  k_gemm_proj<<<gg, 256, 0, stream>>>(xbf, wihf, bfv, gxf);
  k_gemm_proj<<<gg, 256, 0, stream>>>(xbf, wihb, bbv, gxb);

  dim3 gl(Bn / 16, 2);                                        // 32 WGs
  k_lstm<<<gl, 256, LSTM_LDS, stream>>>(gxf, gxb, whhf, whhb, seqlen, hF, hBr);

  k_fc<<<BT / 8, 256, 0, stream>>>(hF, hBr, seqlen, target, fcW, fcb, logits, emitg);
  k_crf<<<Bn / 8, 256, 0, stream>>>(logits, emitg, target, seqlen, trans, startv, endv, out);
}
